// AttentionLayer_57930518888709
// MI455X (gfx1250) — matrix operand order repaired and checked by reading, not compiled, not run on hardware
//
#include <hip/hip_runtime.h>
#include <hip/hip_bf16.h>

typedef __attribute__((ext_vector_type(16))) _Float16 v16h;
typedef __attribute__((ext_vector_type(8)))  float    v8f;

#define F_DIM   64
#define H1D     128
#define H2D     128
#define NSEQ    4096
#define BATCH   64
#define WPB     8      // waves per block (wave32)
#define TPW     8      // 16-token tiles per wave
#define NBLOCKS 256    // 256*8*8*16 = 262144 tokens
#define NEG_BIAS 999.0f

__device__ __forceinline__ unsigned pack2h(float a, float b) {
    union { _Float16 h[2]; unsigned u; } un;
    un.h[0] = (_Float16)a; un.h[1] = (_Float16)b;
    return un.u;
}

// A operand pre-packed in LDS: [tile][lane][8 dwords] -> two contiguous uint4 loads
__device__ __forceinline__ v16h load_tile16(const unsigned* p) {
    union { v16h v; uint4 q[2]; } u;
    u.q[0] = *(const uint4*)(p);
    u.q[1] = *(const uint4*)(p + 4);
    return u.v;
}

// One layer-1 M-tile: c = W1^T[mt] * x^T (2 K-chunks), + bias, ReLU
__device__ __forceinline__ v8f layer1_tile(const unsigned* sW1a, const float* sB1,
                                           int mt, int lane, int h,
                                           v16h b0, v16h b1v) {
    v16h A0 = load_tile16(sW1a + (((mt * 2 + 0) * 32 + lane) << 3));
    v16h A1 = load_tile16(sW1a + (((mt * 2 + 1) * 32 + lane) << 3));
    v8f c = {};
    c = __builtin_amdgcn_wmma_f32_16x16x32_f16(false, A0, false, b0,  (short)0, c, false, false);
    c = __builtin_amdgcn_wmma_f32_16x16x32_f16(false, A1, false, b1v, (short)0, c, false, false);
    const float4* pb = (const float4*)(sB1 + mt * 16 + 8 * h);
    float4 q0 = pb[0], q1 = pb[1];
    float bb[8] = {q0.x, q0.y, q0.z, q0.w, q1.x, q1.y, q1.z, q1.w};
    v8f r;
#pragma unroll
    for (int i = 0; i < 8; ++i) {
        float v = c[i] + bb[i];
        r[i] = v > 0.f ? v : 0.f;
    }
    return r;
}

__global__ __launch_bounds__(WPB * 32)
void mlp_logits_kernel(const float* __restrict__ x, const int* __restrict__ mask,
                       const float* __restrict__ W1, const float* __restrict__ b1,
                       const float* __restrict__ W2, const float* __restrict__ b2,
                       const float* __restrict__ W3, const float* __restrict__ b3,
                       float* __restrict__ logits)
{
    // W1^T as A operand: 8 mt x 2 kc tiles; W2^T as A: 8 mt x 4 kc tiles.
    __shared__ __align__(16) unsigned sW1a[8 * 2 * 32 * 8];   // 16 KB
    __shared__ __align__(16) unsigned sW2a[8 * 4 * 32 * 8];   // 32 KB
    __shared__ __align__(16) float sB1[H1D], sB2[H2D], sW3[H2D];

    const int tid = threadIdx.x;

    // ---- Stage weights: f32 -> f16, transposed + pre-swizzled into A-register layout ----
    for (int i = tid; i < 8 * 2 * 32 * 8; i += WPB * 32) {
        int j = i & 7, ln = (i >> 3) & 31, kc = (i >> 8) & 1, mt = i >> 9;
        int ch = mt * 16 + (ln & 15);                                            // out channel (M)
        int kg = kc * 32 + ((ln >> 4) << 4) + (j << 1);  // in feat (K)
        sW1a[i] = pack2h(W1[kg * H1D + ch], W1[(kg + 1) * H1D + ch]);
    }
    for (int i = tid; i < 8 * 4 * 32 * 8; i += WPB * 32) {
        int j = i & 7, ln = (i >> 3) & 31, kc = (i >> 8) & 3, mt = i >> 10;
        int ch = mt * 16 + (ln & 15);
        int kg = kc * 32 + ((ln >> 4) << 4) + (j << 1);
        sW2a[i] = pack2h(W2[kg * H2D + ch], W2[(kg + 1) * H2D + ch]);
    }
    if (tid < H1D) sB1[tid] = b1[tid];
    if (tid < H2D) { sB2[tid] = b2[tid]; sW3[tid] = W3[tid]; }
    const float b3v = b3[0];
    __syncthreads();   // only barrier in the kernel

    const int wave = tid >> 5, lane = tid & 31;
    const int n = lane & 15, h = lane >> 4;
    const bool hh = (h != 0);

#pragma unroll 1
    for (int t = 0; t < TPW; ++t) {
        // Block LICM: keep the (t-invariant) LDS weight-tile loads inside the loop
        // so they stay ds_load_b128 instead of being hoisted into spilled VGPRs.
        asm volatile("" ::: "memory");

        const int tile = (blockIdx.x * WPB + wave) * TPW + t;
        const int tokBase = tile * 16;

        // ---- B operand for layer 1: x^T tile (64 feat x 16 tokens), f32->f16 ----
        v16h b1r[2];
        const float* xr = x + (size_t)(tokBase + n) * F_DIM;
#pragma unroll
        for (int kc = 0; kc < 2; ++kc) {
            const float4* p = (const float4*)(xr + kc * 32 + h * 16);
            float4 q0 = p[0], q1 = p[1], q2 = p[2], q3 = p[3];
            union { v16h v; unsigned d[8]; } u;
            u.d[0] = pack2h(q0.x, q0.y); u.d[1] = pack2h(q0.z, q0.w);
            u.d[2] = pack2h(q1.x, q1.y); u.d[3] = pack2h(q1.z, q1.w);
            u.d[4] = pack2h(q2.x, q2.y); u.d[5] = pack2h(q2.z, q2.w);
            u.d[6] = pack2h(q3.x, q3.y); u.d[7] = pack2h(q3.z, q3.w);
            b1r[kc] = u.v;
        }

        // ---- Layer 1 fused with C->B conversion: per K-chunk, compute C tiles
        // 2kc & 2kc+1, bias+ReLU, pack f16, half-swap via shfl_xor(16) -> b2r[kc].
        v16h b2r[4];
#pragma unroll
        for (int kc = 0; kc < 4; ++kc) {
            v8f t0 = layer1_tile(sW1a, sB1, 2 * kc,     lane, h, b1r[0], b1r[1]);
            v8f t1 = layer1_tile(sW1a, sB1, 2 * kc + 1, lane, h, b1r[0], b1r[1]);
            unsigned p0[4], p1[4], snd[4], rcv[4];
#pragma unroll
            for (int i = 0; i < 4; ++i) {
                p0[i] = pack2h(t0[2 * i], t0[2 * i + 1]);
                p1[i] = pack2h(t1[2 * i], t1[2 * i + 1]);
            }
#pragma unroll
            for (int i = 0; i < 4; ++i) snd[i] = hh ? p0[i] : p1[i];
#pragma unroll
            for (int i = 0; i < 4; ++i) rcv[i] = __shfl_xor(snd[i], 16, 32);
            union { v16h v; unsigned d[8]; } u;
#pragma unroll
            for (int i = 0; i < 4; ++i) {
                u.d[i]     = hh ? rcv[i] : p0[i];   // K = kc*32 + h*16 + [0,8)
                u.d[i + 4] = hh ? p1[i]  : rcv[i];  // K = kc*32 + h*16 + [8,16)
            }
            b2r[kc] = u.v;
            __builtin_amdgcn_sched_barrier(0);
        }

        // ---- Layer 2 + fused layer 3: partial = sum_ch relu(h2^T[ch]) * W3[ch] ----
        float partial = 0.f;
#pragma unroll
        for (int mt = 0; mt < 8; ++mt) {
            v8f c = {};
#pragma unroll
            for (int kc = 0; kc < 4; ++kc) {
                v16h A = load_tile16(sW2a + (((mt * 4 + kc) * 32 + lane) << 3));
                c = __builtin_amdgcn_wmma_f32_16x16x32_f16(false, A, false, b2r[kc], (short)0, c, false, false);
            }
            const float4* pb = (const float4*)(sB2 + mt * 16 + 8 * h);
            const float4* pw = (const float4*)(sW3 + mt * 16 + 8 * h);
            float4 qb0 = pb[0], qb1 = pb[1], qw0 = pw[0], qw1 = pw[1];
            float bb[8] = {qb0.x, qb0.y, qb0.z, qb0.w, qb1.x, qb1.y, qb1.z, qb1.w};
            float ww[8] = {qw0.x, qw0.y, qw0.z, qw0.w, qw1.x, qw1.y, qw1.z, qw1.w};
#pragma unroll
            for (int r = 0; r < 8; ++r) {
                float v = c[r] + bb[r];
                v = v > 0.f ? v : 0.f;
                partial += v * ww[r];
            }
            __builtin_amdgcn_sched_barrier(0);
        }
        const float total = partial + __shfl_xor(partial, 16, 32);

        // ---- Mask + coalesced store (lanes 0..15 write 16 contiguous tokens) ----
        if (h == 0) {
            const int tok = tokBase + n;
            const float mk = (float)mask[tok];
            logits[tok] = (total + b3v) * mk - NEG_BIAS * (1.0f - mk);
        }
    }
}

__global__ __launch_bounds__(256)
void softmax_kernel(float* __restrict__ io)
{
    __shared__ float red[256];
    const int b = blockIdx.x;
    float* p = io + (size_t)b * NSEQ;
    const int tid = threadIdx.x;

    float vals[16];
    float mx = -1e30f;
#pragma unroll
    for (int i = 0; i < 16; ++i) {
        vals[i] = p[tid + i * 256];
        mx = fmaxf(mx, vals[i]);
    }
    red[tid] = mx; __syncthreads();
    for (int s = 128; s > 0; s >>= 1) {
        if (tid < s) red[tid] = fmaxf(red[tid], red[tid + s]);
        __syncthreads();
    }
    mx = red[0]; __syncthreads();

    float sum = 0.f;
#pragma unroll
    for (int i = 0; i < 16; ++i) {
        vals[i] = __expf(vals[i] - mx);
        sum += vals[i];
    }
    red[tid] = sum; __syncthreads();
    for (int s = 128; s > 0; s >>= 1) {
        if (tid < s) red[tid] += red[tid + s];
        __syncthreads();
    }
    const float inv = 1.0f / red[0];
#pragma unroll
    for (int i = 0; i < 16; ++i) p[tid + i * 256] = vals[i] * inv;
}

extern "C" void kernel_launch(void* const* d_in, const int* in_sizes, int n_in,
                              void* d_out, int out_size, void* d_ws, size_t ws_size,
                              hipStream_t stream)
{
    const float* x    = (const float*)d_in[0];
    const int*   mask = (const int*)  d_in[1];
    const float* W1   = (const float*)d_in[2];
    const float* b1   = (const float*)d_in[3];
    const float* W2   = (const float*)d_in[4];
    const float* b2   = (const float*)d_in[5];
    const float* W3   = (const float*)d_in[6];
    const float* b3   = (const float*)d_in[7];
    float* out = (float*)d_out;

    mlp_logits_kernel<<<NBLOCKS, WPB * 32, 0, stream>>>(x, mask, W1, b1, W2, b2, W3, b3, out);
    softmax_kernel<<<BATCH, 256, 0, stream>>>(out);
}
